// ExpandedQuasiResetableRNN_73821897883846
// MI455X (gfx1250) — matrix operand order repaired and checked by reading, not compiled, not run on hardware
//
#include <hip/hip_runtime.h>

// ---------------------------------------------------------------------------
// Problem constants (from the reference)
// ---------------------------------------------------------------------------
#define B_     16
#define T_     2048
#define D_     256
#define CELL_  512
#define NEIGH_ 7

typedef __bf16 bf16;
typedef __attribute__((ext_vector_type(16))) __bf16 v16bf;
typedef __attribute__((ext_vector_type(8)))  __bf16 bf16x8;
typedef __attribute__((ext_vector_type(4)))  __bf16 bf16x4;
typedef __attribute__((ext_vector_type(8)))  float  v8f;

// Workspace layout (bytes)
static constexpr size_t XB_ELEMS   = (size_t)B_ * T_ * D_;              // 8,388,608
static constexpr size_t XB_BYTES   = XB_ELEMS * 2;                      // 16 MiB
static constexpr size_t WF_PER     = (size_t)NEIGH_ * 8 * 32 * 32 * 16; // 917,504 / filter
static constexpr size_t WF_BYTES   = 2 * WF_PER * 2;                    // both filters, bf16
static constexpr size_t FB_ELEMS   = (size_t)B_ * T_ * CELL_;           // 16,777,216
static constexpr size_t FB_BYTES   = FB_ELEMS * 2;                      // 32 MiB

// ---------------------------------------------------------------------------
// 1) x: f32 -> bf16, 4 elements / thread
// ---------------------------------------------------------------------------
__global__ __launch_bounds__(256) void convert_x_kernel(const float* __restrict__ x,
                                                        bf16* __restrict__ xb) {
    size_t i = ((size_t)blockIdx.x * 256 + threadIdx.x) * 4;
    float4 v = *(const float4*)(x + i);
    bf16x4 o;
    o[0] = (bf16)v.x; o[1] = (bf16)v.y; o[2] = (bf16)v.z; o[3] = (bf16)v.w;
    *(bf16x4*)(xb + i) = o;
}

// ---------------------------------------------------------------------------
// 2) weights: f32 -> bf16 swizzled into WMMA B-fragment order.
//    Fragment layout (contiguous k order used here, 16-bit B 32x16):
//      lane L < 16 : N = L,     K = 0..15  (pairs packed in VGPRs 0..7)
//      lane L >=16 : N = L-16,  K = 16..31
//    Tile id = ((k*8 + kc)*32 + nt); each tile = 32 lanes * 16 bf16 = 1 KB.
// ---------------------------------------------------------------------------
__global__ __launch_bounds__(256) void convert_w_kernel(const float* __restrict__ wz,
                                                        const float* __restrict__ wf,
                                                        bf16* __restrict__ frag) {
    unsigned gid = blockIdx.x * 256 + threadIdx.x;
    if (gid >= 2u * (unsigned)WF_PER) return;
    unsigned filt = gid / (unsigned)WF_PER;
    unsigned r    = gid % (unsigned)WF_PER;
    unsigned e  = r & 15;          // element within lane chunk
    unsigned L  = (r >> 4) & 31;   // lane
    unsigned nt = (r >> 9) & 31;   // n-tile (0..31)
    unsigned kc = (r >> 14) & 7;   // K chunk of 32 within D
    unsigned k  = r >> 17;         // conv tap (0..6)
    unsigned n   = nt * 16 + (L & 15);
    unsigned kin = ((L < 16) ? 0u : 16u) + e;
    unsigned d   = kc * 32 + kin;
    const float* w = filt ? wf : wz;
    frag[gid] = (bf16)w[((size_t)k * D_ + d) * CELL_ + n];
}

// ---------------------------------------------------------------------------
// 3) keep mask: (x[b,t,0] != 0)
// ---------------------------------------------------------------------------
__global__ __launch_bounds__(256) void keep_kernel(const float* __restrict__ x,
                                                   float* __restrict__ keep) {
    int i = blockIdx.x * 256 + threadIdx.x;   // 0 .. B*T-1
    keep[i] = (x[(size_t)i * D_] != 0.0f) ? 1.0f : 0.0f;
}

// ---------------------------------------------------------------------------
// 4) Conv-as-GEMM with v_wmma_f32_16x16x32_bf16, M-blocked for weight reuse.
//    Block = 256 threads = 8 waves; block owns 64 consecutive t rows
//    (4 M-tiles) and 16 of the 64 filter*n-tiles (2 per wave).
//    Each wave: acc[4 M][2 N]; B fragments reused 4x from registers,
//    A fragments reused 2x. Weight L2 traffic: ~1.8 GB (4x less than
//    16-row blocking).
//    LDS: 70 halo'd x rows, stride padded 256->264 bf16 (528 B = 33*16 B)
//    so the 16 lanes of an A fragment hit distinct bank groups.
// ---------------------------------------------------------------------------
#define LDS_ROW   264
#define ROWS_STG  70            // 64 + 6 halo

__global__ __launch_bounds__(256) void conv_wmma_kernel(const bf16* __restrict__ xb,
                                                        const bf16* __restrict__ wfrag,
                                                        float* __restrict__ zout,
                                                        bf16* __restrict__ fout) {
    __shared__ __align__(16) bf16 lds[ROWS_STG * LDS_ROW];

    const int mblk = blockIdx.x;            // 0..511
    const int b    = mblk >> 5;             // 32 m-blocks per batch
    const int t0   = (mblk & 31) << 6;      // 64 rows per block
    const int tid  = threadIdx.x;

    // Stage rows t0-3 .. t0+66 (zero-padded at batch edges), 8 B per copy.
    for (int i = tid; i < ROWS_STG * 64; i += 256) {
        int row = i >> 6, seg = i & 63;
        int t = t0 - 3 + row;
        unsigned long long v = 0ull;
        if (t >= 0 && t < T_)
            v = *(const unsigned long long*)(xb + ((size_t)b * T_ + t) * D_ + seg * 4);
        *(unsigned long long*)(&lds[row * LDS_ROW + seg * 4]) = v;
    }
    __syncthreads();

    const int wave = tid >> 5;
    const int lane = tid & 31;
    // filter*n-tile pair for this wave: nft, nft+1 (same filter: nft is even)
    const int nft  = (blockIdx.y << 4) + (wave << 1);   // 0..63
    const int filt = nft >> 5;                          // 0 = z, 1 = f
    const int nt0  = nft & 31;                          // n-tile of pair[0]

    const bf16* wtile0 = wfrag + (size_t)filt * WF_PER + (size_t)nt0 * 512
                               + (size_t)lane * 16;

    v8f acc[4][2] = {};

    // A-fragment layout (contiguous k order, matching the B packing, 16-bit A 16x32):
    //   lane<16 : M=lane,    chunk0 = K0..7,   chunk1 = K8..15
    //   lane>=16: M=lane-16, chunk0 = K16..23, chunk1 = K24..31
    const int arow = lane & 15;
    const int c0   = (lane < 16) ? 0 : 16;

    for (int q = 0; q < NEIGH_ * 8; ++q) {              // q = k*8 + kc
        const int k  = q >> 3;
        const int kc = q & 7;

        // Load the two B fragments once; reuse across 4 M-tiles.
        const bf16* bp0 = wtile0 + (size_t)q * (32 * 512);
        bf16x8 b0lo = *(const bf16x8*)(bp0);
        bf16x8 b0hi = *(const bf16x8*)(bp0 + 8);
        bf16x8 b1lo = *(const bf16x8*)(bp0 + 512);
        bf16x8 b1hi = *(const bf16x8*)(bp0 + 520);
        v16bf bm0 = __builtin_shufflevector(b0lo, b0hi,
                      0,1,2,3,4,5,6,7,8,9,10,11,12,13,14,15);
        v16bf bm1 = __builtin_shufflevector(b1lo, b1hi,
                      0,1,2,3,4,5,6,7,8,9,10,11,12,13,14,15);

#pragma unroll
        for (int m = 0; m < 4; ++m) {
            const bf16* ap = &lds[(m * 16 + k + arow) * LDS_ROW + kc * 32];
            bf16x8 alo = *(const bf16x8*)(ap + c0);
            bf16x8 ahi = *(const bf16x8*)(ap + c0 + 8);
            v16bf a = __builtin_shufflevector(alo, ahi,
                        0,1,2,3,4,5,6,7,8,9,10,11,12,13,14,15);
            acc[m][0] = __builtin_amdgcn_wmma_f32_16x16x32_bf16(
                            false, a, false, bm0, (short)0, acc[m][0], false, false);
            acc[m][1] = __builtin_amdgcn_wmma_f32_16x16x32_bf16(
                            false, a, false, bm1, (short)0, acc[m][1], false, false);
        }
    }

    // Epilogue: C/D layout -> VGPR r: M = r + 8*(lane>=16), N = lane&15.
    const int col   = lane & 15;
    const int rbase = (lane >> 4) * 8;
#pragma unroll
    for (int j = 0; j < 2; ++j) {
        int c = (nt0 + j) * 16 + col;
#pragma unroll
        for (int m = 0; m < 4; ++m) {
#pragma unroll
            for (int r = 0; r < 8; ++r) {
                int t = t0 + m * 16 + rbase + r;
                float v = acc[m][j][r];
                size_t idx = ((size_t)b * T_ + t) * CELL_ + c;
                if (filt == 0) {
                    zout[idx] = tanhf(v);
                } else {
                    float s = 1.0f / (1.0f + __expf(-v));
                    fout[idx] = (bf16)s;
                }
            }
        }
    }
}

// ---------------------------------------------------------------------------
// 5) Sequential gated scan: one thread per (b, cell) chain.
//    z read in-place from d_out, overwritten with h. Coalesced across c.
// ---------------------------------------------------------------------------
__global__ __launch_bounds__(256) void recurrence_kernel(float* __restrict__ out,
                                                         const bf16* __restrict__ fbuf,
                                                         const float* __restrict__ keep) {
    int gid = blockIdx.x * 256 + threadIdx.x;  // 0 .. B*CELL-1
    int b = gid >> 9;                          // / CELL
    int c = gid & (CELL_ - 1);
    float h = 0.0f;
    size_t base = (size_t)b * T_ * CELL_ + c;
    const float* kp = keep + (size_t)b * T_;
    for (int t = 0; t < T_; ++t) {
        size_t idx = base + (size_t)t * CELL_;
        float z = out[idx];
        float f = (float)fbuf[idx];
        h = (f * h + (1.0f - f) * z) * kp[t];
        out[idx] = h;
    }
}

// ---------------------------------------------------------------------------
// Launcher
// ---------------------------------------------------------------------------
extern "C" void kernel_launch(void* const* d_in, const int* in_sizes, int n_in,
                              void* d_out, int out_size, void* d_ws, size_t ws_size,
                              hipStream_t stream) {
    const float* x  = (const float*)d_in[0];
    const float* fz = (const float*)d_in[1];
    const float* ff = (const float*)d_in[2];
    float* out = (float*)d_out;

    char* ws = (char*)d_ws;
    bf16*  xb    = (bf16*)(ws);
    bf16*  wfrag = (bf16*)(ws + XB_BYTES);
    bf16*  fbuf  = (bf16*)(ws + XB_BYTES + WF_BYTES);
    float* keep  = (float*)(ws + XB_BYTES + WF_BYTES + FB_BYTES);

    convert_x_kernel<<<(int)(XB_ELEMS / 4 / 256), 256, 0, stream>>>(x, xb);
    convert_w_kernel<<<(int)((2 * WF_PER + 255) / 256), 256, 0, stream>>>(fz, ff, wfrag);
    keep_kernel<<<(B_ * T_) / 256, 256, 0, stream>>>(x, keep);

    dim3 ggrid(B_ * (T_ / 64), 4, 1);   // 512 M-blocks x 4 N-groups
    conv_wmma_kernel<<<ggrid, 256, 0, stream>>>(xb, wfrag, out, fbuf);

    recurrence_kernel<<<(B_ * CELL_) / 256, 256, 0, stream>>>(out, fbuf, keep);
}
